// Block_43696997269588
// MI455X (gfx1250) — compile-verified, matrix operand order checked by reading, not run on hardware
//
#include <hip/hip_runtime.h>
#include <hip/hip_bf16.h>

// ---------------------------------------------------------------------------
// Transformer block (pre-LN attention + FFN) for gfx1250 (MI455X).
// All GEMM-shaped math runs on v_wmma_f32_16x16x32_bf16 (fp32 accumulate).
// Weights are converted once to bf16 AND pre-transposed to [N][K] so GEMM
// tile staging is pure 16B vector copies (async global->LDS DMA, ASYNCcnt).
// ---------------------------------------------------------------------------

typedef __attribute__((ext_vector_type(16))) __bf16 v16bf;
typedef __attribute__((ext_vector_type(8)))  float  v8f;
typedef int v4i __attribute__((ext_vector_type(4)));

static constexpr int Bb = 2;
static constexpr int Tt = 2048;
static constexpr int Cc = 1024;
static constexpr int Ff = 4096;
static constexpr int Mm = Bb * Tt;      // 4096 rows
static constexpr float kLnEps = 1e-5f;

// gfx1250 async global->LDS copy path (ASYNCcnt-tracked). Parameter types per
// hipcc diagnostic: (AS1 v4i*, AS3 v4i*, imm offset, imm cpol).
#if __has_builtin(__builtin_amdgcn_global_load_async_to_lds_b128) && \
    __has_builtin(__builtin_amdgcn_s_wait_asynccnt)
#define USE_ASYNC_LDS 1
typedef __attribute__((address_space(1))) v4i g_v4i;
typedef __attribute__((address_space(3))) v4i l_v4i;
#else
#define USE_ASYNC_LDS 0
#endif

union FragAB {
    v16bf v;
    uint4 u[2];
};

__device__ __forceinline__ float gelu_exact(float x) {
    return 0.5f * x * (1.0f + erff(x * 0.70710678118654752440f));
}

// ---------------------------------------------------------------------------
// f32 [K][N] -> bf16 [N][K] transpose-convert (weights, once per call).
// block (32,8), grid (N/32, K/32); LDS-tiled for coalescing both sides.
// ---------------------------------------------------------------------------
__global__ __launch_bounds__(256) void cvt_bf16_t(const float* __restrict__ src,
                                                  __bf16* __restrict__ dst,
                                                  int K, int N) {
    __shared__ float tile[32][33];
    const int n0 = blockIdx.x * 32;
    const int k0 = blockIdx.y * 32;
    const int tx = threadIdx.x, ty = threadIdx.y;
#pragma unroll
    for (int i = 0; i < 32; i += 8)
        tile[ty + i][tx] = src[(size_t)(k0 + ty + i) * N + n0 + tx];
    __syncthreads();
#pragma unroll
    for (int i = 0; i < 32; i += 8)
        dst[(size_t)(n0 + ty + i) * K + k0 + tx] = (__bf16)tile[tx][ty + i];
}

// ---------------------------------------------------------------------------
// Row LayerNorm (fp32 in, bf16 out). One 256-thread block per row of C=1024.
// ---------------------------------------------------------------------------
__global__ __launch_bounds__(256) void ln_bf16(const float* __restrict__ x,
                                               const float* __restrict__ g,
                                               const float* __restrict__ b,
                                               __bf16* __restrict__ out) {
    __shared__ float red[256];
    const int row = blockIdx.x;
    const int tid = threadIdx.x;
    const float* xr = x + (size_t)row * Cc;

    float s = 0.0f;
    for (int i = tid; i < Cc; i += 256) s += xr[i];
    red[tid] = s;
    __syncthreads();
    for (int off = 128; off > 0; off >>= 1) {
        if (tid < off) red[tid] += red[tid + off];
        __syncthreads();
    }
    const float mu = red[0] * (1.0f / Cc);
    __syncthreads();

    float sq = 0.0f;
    for (int i = tid; i < Cc; i += 256) {
        float d = xr[i] - mu;
        sq += d * d;
    }
    red[tid] = sq;
    __syncthreads();
    for (int off = 128; off > 0; off >>= 1) {
        if (tid < off) red[tid] += red[tid + off];
        __syncthreads();
    }
    const float rs = rsqrtf(red[0] * (1.0f / Cc) + kLnEps);

    for (int i = tid; i < Cc; i += 256)
        out[(size_t)row * Cc + i] = (__bf16)((xr[i] - mu) * rs * g[i] + b[i]);
}

// ---------------------------------------------------------------------------
// bf16 GEMM: out[M,N] = A[M,K] @ Wt[N,K]^T (+bias) (+gelu) (+residual)
// A row-major [M][K], Wt pre-transposed [N][K] (both bf16).
// Block tile 128x128, BK=32, 256 threads = 8 waves (2x4 grid), each wave
// owns 64x32 = 4x2 WMMA 16x16 tiles. Double-buffered LDS; async DMA staging
// via global_load_async_to_lds_b128 when available.
// ---------------------------------------------------------------------------
template <bool OUT_BF16, bool HAS_BIAS, bool HAS_RESID, bool DO_GELU>
__global__ __launch_bounds__(256) void gemm_bf16_wmma(
    const __bf16* __restrict__ A, const __bf16* __restrict__ Wt,
    const float* __restrict__ bias, const float* __restrict__ resid,
    void* __restrict__ out, int M, int N, int K) {
    constexpr int AS = 40;  // padded LDS row stride (bf16 elems), 16B-aligned
    __shared__ __bf16 As[2][128 * AS];   // [m][k]
    __shared__ __bf16 Bt[2][128 * AS];   // [n][k]

    const int tid  = threadIdx.x;
    const int bm   = blockIdx.y;
    const int bn   = blockIdx.x;
    const int w    = tid >> 5;
    const int lane = tid & 31;
    const int wm   = w >> 2;          // 0..1
    const int wn   = w & 3;           // 0..3
    const int half = lane >> 4;       // 0..1
    const int lr   = lane & 15;
    const int kb   = half * 8;        // bf16 A/B fragment K base per half-wave

    const v8f vzero = {0.f, 0.f, 0.f, 0.f, 0.f, 0.f, 0.f, 0.f};
    v8f acc[4][2];
#pragma unroll
    for (int mi = 0; mi < 4; mi++)
#pragma unroll
        for (int ni = 0; ni < 2; ni++) acc[mi][ni] = vzero;

    // stage one 128x32 A tile and one 128x32 W^T tile into LDS buffer `buf`
    auto stage = [&](int k0, int buf) {
        for (int i = tid; i < 512; i += 256) {
            const int r = i >> 2;           // row (m or n), 0..127
            const int c = (i & 3) * 8;      // k offset, 0/8/16/24
#if USE_ASYNC_LDS
            __builtin_amdgcn_global_load_async_to_lds_b128(
                (g_v4i*)&A[(size_t)(bm * 128 + r) * K + k0 + c],
                (l_v4i*)&As[buf][r * AS + c], 0, 0);
            __builtin_amdgcn_global_load_async_to_lds_b128(
                (g_v4i*)&Wt[(size_t)(bn * 128 + r) * K + k0 + c],
                (l_v4i*)&Bt[buf][r * AS + c], 0, 0);
#else
            *(uint4*)&As[buf][r * AS + c] =
                *(const uint4*)&A[(size_t)(bm * 128 + r) * K + k0 + c];
            *(uint4*)&Bt[buf][r * AS + c] =
                *(const uint4*)&Wt[(size_t)(bn * 128 + r) * K + k0 + c];
#endif
        }
    };

    const int ntiles = K >> 5;
    stage(0, 0);
#if USE_ASYNC_LDS
    __builtin_amdgcn_s_wait_asynccnt(0);
#endif
    __syncthreads();

    for (int kt = 0; kt < ntiles; kt++) {
        const int cur = kt & 1;
        if (kt + 1 < ntiles) {
            stage((kt + 1) << 5, cur ^ 1);
        }
#if !USE_ASYNC_LDS
        else {
            // nothing to stage; nudge next C-tile rows toward the caches
            __builtin_prefetch(&A[(size_t)(bm * 128 + (tid & 127)) * K]);
        }
#endif
        FragAB af[4], bf[2];
#pragma unroll
        for (int mi = 0; mi < 4; mi++) {
            const int ar = wm * 64 + mi * 16 + lr;
            af[mi].u[0] = *(const uint4*)&As[cur][ar * AS + kb];
            af[mi].u[1] = *(const uint4*)&As[cur][ar * AS + kb + 16];
        }
#pragma unroll
        for (int ni = 0; ni < 2; ni++) {
            const int bc = wn * 32 + ni * 16 + lr;
            bf[ni].u[0] = *(const uint4*)&Bt[cur][bc * AS + kb];
            bf[ni].u[1] = *(const uint4*)&Bt[cur][bc * AS + kb + 16];
        }
#pragma unroll
        for (int mi = 0; mi < 4; mi++)
#pragma unroll
            for (int ni = 0; ni < 2; ni++)
                acc[mi][ni] = __builtin_amdgcn_wmma_f32_16x16x32_bf16(
                    false, af[mi].v, false, bf[ni].v, (short)0, acc[mi][ni],
                    false, false);
#if USE_ASYNC_LDS
        __builtin_amdgcn_s_wait_asynccnt(0);
#endif
        __syncthreads();
    }

    // epilogue: C/D layout -> VGPR r holds (M=r, N=lane) lanes 0-15,
    // (M=8+r, N=lane-16) lanes 16-31.
#pragma unroll
    for (int mi = 0; mi < 4; mi++) {
#pragma unroll
        for (int ni = 0; ni < 2; ni++) {
            const int n = bn * 128 + wn * 32 + ni * 16 + lr;
#pragma unroll
            for (int r = 0; r < 8; r++) {
                const int m = bm * 128 + wm * 64 + mi * 16 + r + half * 8;
                float vv = acc[mi][ni][r];
                if (HAS_BIAS) vv += bias[n];
                if (DO_GELU) vv = gelu_exact(vv);
                if (HAS_RESID) vv += resid[(size_t)m * N + n];
                if (OUT_BF16)
                    ((__bf16*)out)[(size_t)m * N + n] = (__bf16)vv;
                else
                    ((float*)out)[(size_t)m * N + n] = vv;
            }
        }
    }
}

// ---------------------------------------------------------------------------
// Flash attention, head dim 64. One block = one (b,h) x 64-query tile.
// S = (Q K^T)/8 via WMMA; online softmax in LDS; O += P V via WMMA.
// ---------------------------------------------------------------------------
__global__ __launch_bounds__(256) void attn_wmma(const __bf16* __restrict__ q,
                                                 const __bf16* __restrict__ k,
                                                 const __bf16* __restrict__ v,
                                                 __bf16* __restrict__ o) {
    constexpr int ST = 72;  // padded bf16 stride for 64-wide tiles
    __shared__ __bf16 Qs[64 * ST];    // [t][d]
    __shared__ __bf16 KPs[64 * ST];   // K tile [s][d], later reused as P [t][s]
    __shared__ __bf16 Vt[64 * ST];    // V transposed [d][s]
    __shared__ float  Ss[64 * 64];    // scores (fp32)
    __shared__ float  Os[64 * 64];    // output accumulator (fp32)
    __shared__ float  rowm[64], rowl[64];

    const int tid  = threadIdx.x;
    const int bh   = blockIdx.x;
    const int bq   = blockIdx.y;
    const int bb   = bh >> 4;
    const int hh   = bh & 15;
    const size_t hbase = ((size_t)bb * Tt) * Cc + (size_t)hh * 64;

    const int w = tid >> 5, lane = tid & 31;
    const int half = lane >> 4, lr = lane & 15, kb = half * 8;
    const int sm = w & 3;   // row group 0..3
    const int sn = w >> 2;  // col super-group 0..1
    const v8f vzero = {0.f, 0.f, 0.f, 0.f, 0.f, 0.f, 0.f, 0.f};

    // load Q tile
    for (int i = tid; i < 512; i += 256) {
        const int r = i >> 3, c = (i & 7) * 8;
        *(uint4*)&Qs[r * ST + c] =
            *(const uint4*)&q[hbase + (size_t)(bq * 64 + r) * Cc + c];
    }
    for (int i = tid; i < 64 * 64; i += 256) Os[i] = 0.0f;
    if (tid < 64) { rowm[tid] = -1e30f; rowl[tid] = 0.0f; }

    for (int s0 = 0; s0 < Tt; s0 += 64) {
        // stage K tile [s][d]
        for (int i = tid; i < 512; i += 256) {
            const int r = i >> 3, c = (i & 7) * 8;
            *(uint4*)&KPs[r * ST + c] =
                *(const uint4*)&k[hbase + (size_t)(s0 + r) * Cc + c];
        }
        // stage V tile transposed [d][s]
        for (int i = tid; i < 512; i += 256) {
            const int r = i >> 3, c = (i & 7) * 8;
            uint4 vv = *(const uint4*)&v[hbase + (size_t)(s0 + r) * Cc + c];
            const __bf16* p = (const __bf16*)&vv;
#pragma unroll
            for (int j = 0; j < 8; j++) Vt[(c + j) * ST + r] = p[j];
        }
        __syncthreads();

        // ---- S = Q K^T (each wave: 1 row group x 2 col tiles) ----
        v8f sacc[2] = {vzero, vzero};
#pragma unroll
        for (int kk = 0; kk < 64; kk += 32) {
            FragAB af;
            const int ar = sm * 16 + lr;
            af.u[0] = *(const uint4*)&Qs[ar * ST + kk + kb];
            af.u[1] = *(const uint4*)&Qs[ar * ST + kk + kb + 16];
#pragma unroll
            for (int ni = 0; ni < 2; ni++) {
                FragAB bf;
                const int bc = sn * 32 + ni * 16 + lr;  // key index
                bf.u[0] = *(const uint4*)&KPs[bc * ST + kk + kb];
                bf.u[1] = *(const uint4*)&KPs[bc * ST + kk + kb + 16];
                sacc[ni] = __builtin_amdgcn_wmma_f32_16x16x32_bf16(
                    false, af.v, false, bf.v, (short)0, sacc[ni], false, false);
            }
        }
#pragma unroll
        for (int ni = 0; ni < 2; ni++) {
            const int cn = sn * 32 + ni * 16 + lr;
#pragma unroll
            for (int r = 0; r < 8; r++) {
                const int rm = sm * 16 + r + half * 8;
                Ss[rm * 64 + cn] = sacc[ni][r] * 0.125f;  // 1/sqrt(64)
            }
        }
        __syncthreads();

        // ---- online softmax; write P (bf16) into KPs ----
        if (tid < 64) {
            const int r = tid;
            const float mold = rowm[r];
            float mx = mold;
            for (int j = 0; j < 64; j++) mx = fmaxf(mx, Ss[r * 64 + j]);
            const float alpha = __expf(mold - mx);
            float l = rowl[r] * alpha;
            for (int j = 0; j < 64; j++) {
                const float p = __expf(Ss[r * 64 + j] - mx);
                l += p;
                KPs[r * ST + j] = (__bf16)p;
            }
            for (int j = 0; j < 64; j++) Os[r * 64 + j] *= alpha;
            rowm[r] = mx;
            rowl[r] = l;
        }
        __syncthreads();

        // ---- O += P V ----
        v8f oacc[2] = {vzero, vzero};
#pragma unroll
        for (int kk = 0; kk < 64; kk += 32) {
            FragAB af;
            const int ar = sm * 16 + lr;   // query row
            af.u[0] = *(const uint4*)&KPs[ar * ST + kk + kb];
            af.u[1] = *(const uint4*)&KPs[ar * ST + kk + kb + 16];
#pragma unroll
            for (int ni = 0; ni < 2; ni++) {
                FragAB bf;
                const int bc = sn * 32 + ni * 16 + lr;  // d column
                bf.u[0] = *(const uint4*)&Vt[bc * ST + kk + kb];
                bf.u[1] = *(const uint4*)&Vt[bc * ST + kk + kb + 16];
                oacc[ni] = __builtin_amdgcn_wmma_f32_16x16x32_bf16(
                    false, af.v, false, bf.v, (short)0, oacc[ni], false, false);
            }
        }
#pragma unroll
        for (int ni = 0; ni < 2; ni++) {
            const int cn = sn * 32 + ni * 16 + lr;
#pragma unroll
            for (int r = 0; r < 8; r++) {
                const int rm = sm * 16 + r + half * 8;
                Os[rm * 64 + cn] += oacc[ni][r];
            }
        }
        __syncthreads();
    }

    // finalize: divide by softmax denominator, emit bf16
    if (tid < 64) {
        const int r = tid;
        const float inv = 1.0f / rowl[r];
        for (int j = 0; j < 64; j++)
            o[hbase + (size_t)(bq * 64 + r) * Cc + j] =
                (__bf16)(Os[r * 64 + j] * inv);
    }
}

// ---------------------------------------------------------------------------
// Host-side orchestration
// ---------------------------------------------------------------------------
extern "C" void kernel_launch(void* const* d_in, const int* in_sizes, int n_in,
                              void* d_out, int out_size, void* d_ws,
                              size_t ws_size, hipStream_t stream) {
    (void)in_sizes; (void)n_in; (void)out_size; (void)ws_size;
    const float* x   = (const float*)d_in[0];
    const float* Wq  = (const float*)d_in[1];
    const float* Wk  = (const float*)d_in[2];
    const float* Wv  = (const float*)d_in[3];
    const float* Wp  = (const float*)d_in[4];
    const float* bp  = (const float*)d_in[5];
    const float* W1  = (const float*)d_in[6];
    const float* b1  = (const float*)d_in[7];
    const float* W2  = (const float*)d_in[8];
    const float* b2  = (const float*)d_in[9];
    const float* g1  = (const float*)d_in[10];
    const float* be1 = (const float*)d_in[11];
    const float* g2  = (const float*)d_in[12];
    const float* be2 = (const float*)d_in[13];
    float* out = (float*)d_out;

    char* ws = (char*)d_ws;
    size_t off = 0;
    auto alloc = [&](size_t bytes) -> char* {
        char* p = ws + off;
        off += (bytes + 255) & ~size_t(255);
        return p;
    };
    // pre-transposed bf16 weights: Wt[N][K]
    __bf16* Wq_t = (__bf16*)alloc((size_t)Cc * Cc * 2);
    __bf16* Wk_t = (__bf16*)alloc((size_t)Cc * Cc * 2);
    __bf16* Wv_t = (__bf16*)alloc((size_t)Cc * Cc * 2);
    __bf16* Wp_t = (__bf16*)alloc((size_t)Cc * Cc * 2);
    __bf16* W1_t = (__bf16*)alloc((size_t)Cc * Ff * 2);
    __bf16* W2_t = (__bf16*)alloc((size_t)Ff * Cc * 2);
    __bf16* h_b  = (__bf16*)alloc((size_t)Mm * Cc * 2);
    __bf16* q_b  = (__bf16*)alloc((size_t)Mm * Cc * 2);
    __bf16* k_b  = (__bf16*)alloc((size_t)Mm * Cc * 2);
    __bf16* v_b  = (__bf16*)alloc((size_t)Mm * Cc * 2);
    __bf16* o_b  = (__bf16*)alloc((size_t)Mm * Cc * 2);
    float*  x1   = (float*) alloc((size_t)Mm * Cc * 4);
    __bf16* h2_b = (__bf16*)alloc((size_t)Mm * Cc * 2);
    __bf16* f1_b = (__bf16*)alloc((size_t)Mm * Ff * 2);

    // 1) weights -> bf16, transposed to [N][K]
    const dim3 tb(32, 8);
    cvt_bf16_t<<<dim3(Cc / 32, Cc / 32), tb, 0, stream>>>(Wq, Wq_t, Cc, Cc);
    cvt_bf16_t<<<dim3(Cc / 32, Cc / 32), tb, 0, stream>>>(Wk, Wk_t, Cc, Cc);
    cvt_bf16_t<<<dim3(Cc / 32, Cc / 32), tb, 0, stream>>>(Wv, Wv_t, Cc, Cc);
    cvt_bf16_t<<<dim3(Cc / 32, Cc / 32), tb, 0, stream>>>(Wp, Wp_t, Cc, Cc);
    cvt_bf16_t<<<dim3(Ff / 32, Cc / 32), tb, 0, stream>>>(W1, W1_t, Cc, Ff);
    cvt_bf16_t<<<dim3(Cc / 32, Ff / 32), tb, 0, stream>>>(W2, W2_t, Ff, Cc);

    // 2) LN1
    ln_bf16<<<Mm, 256, 0, stream>>>(x, g1, be1, h_b);

    // 3) Q/K/V projections
    dim3 gqkv(Cc / 128, Mm / 128);
    gemm_bf16_wmma<true, false, false, false>
        <<<gqkv, 256, 0, stream>>>(h_b, Wq_t, nullptr, nullptr, q_b, Mm, Cc, Cc);
    gemm_bf16_wmma<true, false, false, false>
        <<<gqkv, 256, 0, stream>>>(h_b, Wk_t, nullptr, nullptr, k_b, Mm, Cc, Cc);
    gemm_bf16_wmma<true, false, false, false>
        <<<gqkv, 256, 0, stream>>>(h_b, Wv_t, nullptr, nullptr, v_b, Mm, Cc, Cc);

    // 4) attention
    attn_wmma<<<dim3(Bb * 16, Tt / 64), 256, 0, stream>>>(q_b, k_b, v_b, o_b);

    // 5) output projection + residual (fp32 out)
    gemm_bf16_wmma<false, true, true, false>
        <<<gqkv, 256, 0, stream>>>(o_b, Wp_t, bp, x, x1, Mm, Cc, Cc);

    // 6) LN2
    ln_bf16<<<Mm, 256, 0, stream>>>(x1, g2, be2, h2_b);

    // 7) FFN up + GELU (bf16 out)
    gemm_bf16_wmma<true, true, false, true>
        <<<dim3(Ff / 128, Mm / 128), 256, 0, stream>>>(h2_b, W1_t, b1, nullptr,
                                                       f1_b, Mm, Ff, Cc);

    // 8) FFN down + residual -> d_out (fp32)
    gemm_bf16_wmma<false, true, true, false>
        <<<gqkv, 256, 0, stream>>>(f1_b, W2_t, b2, x1, out, Mm, Cc, Ff);
}
